// Q2A_7559142441028
// MI455X (gfx1250) — matrix operand order repaired and checked by reading, not compiled, not run on hardware
//
#include <hip/hip_runtime.h>

// ---------------------------------------------------------------------------
// Model dims
// ---------------------------------------------------------------------------
enum : int { cD = 768, cM = 5, cDS = 768, cNH = 12, cB = 8, cV = 512,
             cS = 128, cT = 16, cA = 8, cDIN = 6144 /* D*(3+M) */ };

typedef __attribute__((ext_vector_type(16))) __bf16 v16bf;
typedef __attribute__((ext_vector_type(8)))  float  v8f;

// fp32 -> bf16 (round to nearest even)
__device__ __forceinline__ unsigned short f2bf(float f) {
    unsigned u = __builtin_bit_cast(unsigned, f);
    unsigned r = u + 0x7FFFu + ((u >> 16) & 1u);
    return (unsigned short)(r >> 16);
}
__device__ __forceinline__ unsigned short to_bf_bits(float v)          { return f2bf(v); }
__device__ __forceinline__ unsigned short to_bf_bits(unsigned short v) { return v; }

// two 16B LDS chunks -> one 32B bf16 fragment (2x ds_load_b128)
__device__ __forceinline__ v16bf load_frag16(const unsigned short* p0,
                                             const unsigned short* p1) {
    struct P { uint4 a, b; } pr;
    pr.a = *(const uint4*)p0;
    pr.b = *(const uint4*)p1;
    return __builtin_bit_cast(v16bf, pr);
}
__device__ __forceinline__ uint4 pack8bf(float4 f0, float4 f1) {
    uint4 p;
    p.x = (unsigned)f2bf(f0.x) | ((unsigned)f2bf(f0.y) << 16);
    p.y = (unsigned)f2bf(f0.z) | ((unsigned)f2bf(f0.w) << 16);
    p.z = (unsigned)f2bf(f1.x) | ((unsigned)f2bf(f1.y) << 16);
    p.w = (unsigned)f2bf(f1.z) | ((unsigned)f2bf(f1.w) << 16);
    return p;
}

// ---------------------------------------------------------------------------
// Generic batched WMMA GEMM:  C = act(alpha * A @ op(B) + bias)
//   bKxN==0 -> B is weights (N x K row-major, A@B^T); fp32 or bf16-bits
//   bKxN==1 -> B is (K x N row-major, A@B); fp32
//   Block 256 thr = 8 waves; tile 64x128; wave = 32x32 (4 accumulators,
//   4 v_wmma per K-chunk).  Double-buffered LDS, software-pipelined global
//   loads, one barrier per K-chunk.  LDS rows padded to 40 ushorts (80B) so
//   every lane fragment is two aligned 16B chunks -> ds_load_b128.
// ---------------------------------------------------------------------------
#define BM 64
#define BN 128
#define BKC 32
#define LP 40

template <typename BT>
__global__ __launch_bounds__(256)
void gemm_wmma(const float* __restrict__ A, long lda, long sAb, long sAh,
               const BT* __restrict__ Bm, long ldb, long sBb, long sBh, int bKxN,
               float* __restrict__ C, long ldc, long sCb, long sCh,
               const float* __restrict__ bias,
               int M, int N, int K, int nh, float alpha, int act)
{
    __shared__ unsigned short Alds[2][BM][LP];   // [buf][m][k]
    __shared__ unsigned short Blds[2][BN][LP];   // [buf][n][k]

    const int z  = blockIdx.z;
    const int bb = z / nh;
    const int hh = z - bb * nh;
    const float* Ab = A  + (long)bb * sAb + (long)hh * sAh;
    const BT*    Bb = Bm + (long)bb * sBb + (long)hh * sBh;
    float*       Cb = C  + (long)bb * sCb + (long)hh * sCh;

    const int m0 = blockIdx.x * BM;
    const int n0 = blockIdx.y * BN;
    const int tid  = threadIdx.x;
    const int lane = tid & 31;
    const int wave = tid >> 5;
    const int mwave = (wave & 1) << 5;   // 0,32
    const int nwave = (wave >> 1) << 5;  // 0,32,64,96
    const int half = lane >> 4;
    const int l15  = lane & 15;

    // fast-path staging coordinates
    const int ar  = tid >> 2;          // 0..63   A row
    const int ac  = (tid & 3) << 3;    // 0..24   A k-chunk (8)
    const int br  = tid >> 1;          // 0..127  B row (W-style)
    const int bc  = (tid & 1) << 4;    // 0,16    B k-chunk (16)
    const int bkr = tid >> 3;          // 0..31   B k-row (KxN)
    const int bnc = (tid & 7) << 4;    // 0..112  B n-chunk (16, KxN)

    v8f acc00 = {}, acc01 = {}, acc10 = {}, acc11 = {};

    // staging registers for the pipelined path
    float4 fa0, fa1;
    float4 fb0, fb1, fb2, fb3;
    uint4  ub0, ub1;

    auto loadA = [&](int kk) {
        const float* s = Ab + (long)(m0 + ar) * lda + (kk + ac);
        fa0 = *(const float4*)s;
        fa1 = *(const float4*)(s + 4);
    };
    auto storeA = [&](int buf) {
        *(uint4*)&Alds[buf][ar][ac] = pack8bf(fa0, fa1);
    };
    auto loadB = [&](int kk) {
        if (bKxN) {
            const float* s = (const float*)Bb + (long)(kk + bkr) * ldb + (n0 + bnc);
            fb0 = *(const float4*)s;       fb1 = *(const float4*)(s + 4);
            fb2 = *(const float4*)(s + 8); fb3 = *(const float4*)(s + 12);
        } else if constexpr (sizeof(BT) == 2) {
            const BT* s = Bb + (long)(n0 + br) * ldb + (kk + bc);
            ub0 = *(const uint4*)s;
            ub1 = *(const uint4*)(s + 8);
            if (kk + BKC < K) __builtin_prefetch(s + BKC, 0, 1);
        } else {
            const float* s = (const float*)Bb + (long)(n0 + br) * ldb + (kk + bc);
            fb0 = *(const float4*)s;       fb1 = *(const float4*)(s + 4);
            fb2 = *(const float4*)(s + 8); fb3 = *(const float4*)(s + 12);
        }
    };
    auto storeB = [&](int buf) {
        if (bKxN) {
            unsigned short h[16] = { f2bf(fb0.x), f2bf(fb0.y), f2bf(fb0.z), f2bf(fb0.w),
                                     f2bf(fb1.x), f2bf(fb1.y), f2bf(fb1.z), f2bf(fb1.w),
                                     f2bf(fb2.x), f2bf(fb2.y), f2bf(fb2.z), f2bf(fb2.w),
                                     f2bf(fb3.x), f2bf(fb3.y), f2bf(fb3.z), f2bf(fb3.w) };
#pragma unroll
            for (int j = 0; j < 16; ++j) Blds[buf][bnc + j][bkr] = h[j];
        } else if constexpr (sizeof(BT) == 2) {
            *(uint4*)&Blds[buf][br][bc]     = ub0;
            *(uint4*)&Blds[buf][br][bc + 8] = ub1;
        } else {
            *(uint4*)&Blds[buf][br][bc]     = pack8bf(fb0, fb1);
            *(uint4*)&Blds[buf][br][bc + 8] = pack8bf(fb2, fb3);
        }
    };
    auto compute = [&](int buf) {
        const unsigned short* a0r = &Alds[buf][mwave + l15][0];
        const unsigned short* a1r = &Alds[buf][mwave + 16 + l15][0];
        v16bf a0 = load_frag16(a0r + half * 8, a0r + 16 + half * 8);
        v16bf a1 = load_frag16(a1r + half * 8, a1r + 16 + half * 8);
        const unsigned short* b0r = &Blds[buf][nwave + l15][0];
        const unsigned short* b1r = &Blds[buf][nwave + 16 + l15][0];
        v16bf b0 = load_frag16(b0r + half * 8, b0r + 16 + half * 8);
        v16bf b1 = load_frag16(b1r + half * 8, b1r + 16 + half * 8);
        acc00 = __builtin_amdgcn_wmma_f32_16x16x32_bf16(false, a0, false, b0, (short)0, acc00, false, false);
        acc01 = __builtin_amdgcn_wmma_f32_16x16x32_bf16(false, a0, false, b1, (short)0, acc01, false, false);
        acc10 = __builtin_amdgcn_wmma_f32_16x16x32_bf16(false, a1, false, b0, (short)0, acc10, false, false);
        acc11 = __builtin_amdgcn_wmma_f32_16x16x32_bf16(false, a1, false, b1, (short)0, acc11, false, false);
    };

    const bool fast = (m0 + BM <= M) && (n0 + BN <= N) && ((K & (BKC - 1)) == 0);

    if (fast) {
        // -------- software-pipelined, double-buffered main path --------
        loadA(0); loadB(0);
        storeA(0); storeB(0);
        __syncthreads();
        int cur = 0;
        for (int kk = 0; kk < K; kk += BKC) {
            const bool hn = (kk + BKC) < K;
            if (hn) { loadA(kk + BKC); loadB(kk + BKC); }   // hide under WMMA
            compute(cur);
            if (hn) { storeA(cur ^ 1); storeB(cur ^ 1); }
            __syncthreads();
            cur ^= 1;
        }
    } else {
        // -------- edge path: clamped-index loads (no exec branching) --------
        for (int kk = 0; kk < K; kk += BKC) {
            for (int e = tid; e < BM * BKC; e += 256) {
                int r = e >> 5, c = e & 31;
                int gm = m0 + r, gk = kk + c;
                int cm = gm < M ? gm : M - 1;
                int ck = gk < K ? gk : K - 1;
                float v = Ab[(long)cm * lda + ck];
                Alds[0][r][c] = (gm < M && gk < K) ? f2bf(v) : (unsigned short)0;
            }
            for (int e = tid; e < BN * BKC; e += 256) {
                int nn, kc;
                if (bKxN) { kc = e >> 7; nn = e & 127; }
                else      { nn = e >> 5; kc = e & 31; }
                int gn = n0 + nn, gk = kk + kc;
                int cn = gn < N ? gn : N - 1;
                int ck = gk < K ? gk : K - 1;
                unsigned short v;
                if (bKxN) v = to_bf_bits(((const float*)Bb)[(long)ck * ldb + cn]);
                else      v = to_bf_bits(Bb[(long)cn * ldb + ck]);
                Blds[0][nn][kc] = (gn < N && gk < K) ? v : (unsigned short)0;
            }
            __syncthreads();
            compute(0);
            __syncthreads();
        }
    }

    // ---- epilogue: C layout m = i + 8*half, n = lane&15 ----
    auto store_acc = [&](const v8f& acc, int mt, int nt) {
#pragma unroll
        for (int i = 0; i < 8; ++i) {
            int gm = m0 + mwave + mt * 16 + half * 8 + i;
            int gn = n0 + nwave + nt * 16 + l15;
            if (gm < M && gn < N) {
                float v = acc[i] * alpha;
                if (bias) v += bias[gn];
                if (act == 1) v = fmaxf(v, 0.0f);
                Cb[(long)gm * ldc + gn] = v;
            }
        }
    };
    store_acc(acc00, 0, 0); store_acc(acc01, 0, 1);
    store_acc(acc10, 1, 0); store_acc(acc11, 1, 1);
}

// ---------------------------------------------------------------------------
// Small helper kernels
// ---------------------------------------------------------------------------
__global__ void k_f32_to_bf16(const float* __restrict__ in,
                              unsigned short* __restrict__ out, long n) {
    long i = (long)blockIdx.x * 256 + threadIdx.x;
    if (i < n) out[i] = f2bf(in[i]);
}

__global__ __launch_bounds__(128)
void k_softmax_rows(float* __restrict__ data, int cols) {
    __shared__ float red[128];
    long row = blockIdx.x;
    float* p = data + row * (long)cols;
    int tid = threadIdx.x;
    float mx = -3.0e38f;
    for (int c = tid; c < cols; c += 128) mx = fmaxf(mx, p[c]);
    red[tid] = mx; __syncthreads();
    for (int off = 64; off; off >>= 1) {
        if (tid < off) red[tid] = fmaxf(red[tid], red[tid + off]);
        __syncthreads();
    }
    mx = red[0]; __syncthreads();
    float sum = 0.0f;
    for (int c = tid; c < cols; c += 128) { float e = __expf(p[c] - mx); p[c] = e; sum += e; }
    red[tid] = sum; __syncthreads();
    for (int off = 64; off; off >>= 1) {
        if (tid < off) red[tid] += red[tid + off];
        __syncthreads();
    }
    float inv = 1.0f / red[0];
    for (int c = tid; c < cols; c += 128) p[c] *= inv;
}

// wmask[b,a,s] = mean_h attn2[b,h,a,s]
__global__ void k_wmask_mean(const float* __restrict__ s2, float* __restrict__ wm) {
    long i = (long)blockIdx.x * 256 + threadIdx.x;
    const long n = (long)cB * cA * cS;
    if (i >= n) return;
    int s = i % cS; long ba = i / cS; int a = ba % cA; int b = ba / cA;
    float acc = 0.0f;
    for (int h = 0; h < cNH; ++h)
        acc += s2[(((long)b * cNH + h) * cA + a) * cS + s];
    wm[i] = acc * (1.0f / cNH);
}

// scatter a_b (B*A*M, D) -> inputs[(b*A+a)*DIN + 2304 + m*D + c]
__global__ void k_scatter_ab(const float* __restrict__ src, float* __restrict__ inputs) {
    long i = (long)blockIdx.x * 256 + threadIdx.x;
    const long n = (long)cB * cA * cM * cD;
    if (i >= n) return;
    int c = i % cD; long r = i / cD;
    int m = r % cM; long ba = r / cM;
    inputs[ba * cDIN + 3 * cD + (long)m * cD + c] = src[i];
}

// inputs[.., 1536 + c] = q_t[b,c] + a_t[b,a,c]
__global__ void k_qa_add(const float* __restrict__ q_t, const float* __restrict__ a_t,
                         float* __restrict__ inputs) {
    long i = (long)blockIdx.x * 256 + threadIdx.x;
    const long n = (long)cB * cA * cD;
    if (i >= n) return;
    int c = i % cD; long ba = i / cD; int b = ba / cA;
    inputs[ba * cDIN + 2 * cD + c] = q_t[(long)b * cD + c] + a_t[i];
}

__global__ void k_bcast_state(const float* __restrict__ s0, float* __restrict__ st) {
    long i = (long)blockIdx.x * 256 + threadIdx.x;
    if (i >= (long)cB * cDS) return;
    st[i] = s0[i % cDS];
}

// GRU elementwise: states (B*A, DS)
__global__ void k_gru(const float* __restrict__ gi, const float* __restrict__ gh,
                      const float* __restrict__ state, float* __restrict__ states) {
    long i = (long)blockIdx.x * 256 + threadIdx.x;
    const long n = (long)cB * cA * cDS;
    if (i >= n) return;
    int c = i % cDS; long row = i / cDS; int b = row / cA;
    const float* gir = gi + row * (long)(3 * cDS);
    const float* ghr = gh + (long)b * (3 * cDS);
    float ir = gir[c], iz = gir[cDS + c], inn = gir[2 * cDS + c];
    float hr = ghr[c], hz = ghr[cDS + c], hn  = ghr[2 * cDS + c];
    float h  = state[(long)b * cDS + c];
    float r  = 1.0f / (1.0f + __expf(-(ir + hr)));
    float zz = 1.0f / (1.0f + __expf(-(iz + hz)));
    float nn = tanhf(inn + r * hn);
    states[i] = (1.0f - zz) * nn + zz * h;
}

// logits = proj_mid @ w2 + b2 ; write out[b,t,:] ; state[b] = pre[b, argmax]
__global__ __launch_bounds__(256)
void k_logits_argmax(const float* __restrict__ proj_mid, const float* __restrict__ w2,
                     const float* __restrict__ b2, const float* __restrict__ pre,
                     float* __restrict__ out, float* __restrict__ state, int t) {
    int b = blockIdx.x;
    int tid = threadIdx.x, lane = tid & 31, wave = tid >> 5;  // 8 waves = A answers
    __shared__ float lg[cA];
    __shared__ int amax_s;
    const float* row = proj_mid + ((long)b * cA + wave) * cDS;
    float s = 0.0f;
    for (int c = lane; c < cDS; c += 32) s += row[c] * w2[c];
    for (int off = 16; off; off >>= 1) s += __shfl_xor(s, off, 32);
    if (lane == 0) lg[wave] = s + b2[0];
    __syncthreads();
    if (tid == 0) {
        int am = 0; float best = lg[0];
        for (int a = 1; a < cA; ++a) if (lg[a] > best) { best = lg[a]; am = a; }
        amax_s = am;
        for (int a = 0; a < cA; ++a)
            out[((long)b * cT + t) * cA + a] = lg[a];
    }
    __syncthreads();
    const float* src = pre + ((long)b * cA + amax_s) * cDS;
    float* dst = state + (long)b * cDS;
    for (int c = tid; c < cDS; c += 256) dst[c] = src[c];
}

// ---------------------------------------------------------------------------
// Host-side launch helpers
// ---------------------------------------------------------------------------
static inline void gemm(hipStream_t st,
                        const float* A, long lda, long sAb, long sAh,
                        const void* Bm, bool bf16B, long ldb, long sBb, long sBh, int bKxN,
                        float* C, long ldc, long sCb, long sCh,
                        const float* bias, int M, int N, int K,
                        int batch, int nh, float alpha, int act)
{
    dim3 grid((M + BM - 1) / BM, (N + BN - 1) / BN, batch);
    if (bf16B)
        gemm_wmma<unsigned short><<<grid, 256, 0, st>>>(
            A, lda, sAb, sAh, (const unsigned short*)Bm, ldb, sBb, sBh, bKxN,
            C, ldc, sCb, sCh, bias, M, N, K, nh, alpha, act);
    else
        gemm_wmma<float><<<grid, 256, 0, st>>>(
            A, lda, sAb, sAh, (const float*)Bm, ldb, sBb, sBh, bKxN,
            C, ldc, sCb, sCh, bias, M, N, K, nh, alpha, act);
}
static inline int gcover(long n) { return (int)((n + 255) / 256); }

extern "C" void kernel_launch(void* const* d_in, const int* /*in_sizes*/, int /*n_in*/,
                              void* d_out, int /*out_size*/, void* d_ws, size_t /*ws_size*/,
                              hipStream_t stream)
{
    const float* video     = (const float*)d_in[0];
    const float* script    = (const float*)d_in[1];
    const float* question  = (const float*)d_in[2];
    const float* a_texts   = (const float*)d_in[3];
    const float* a_buttons = (const float*)d_in[4];
    const float* state0    = (const float*)d_in[5];
    const float* v_w1 = (const float*)d_in[6];  const float* v_b1 = (const float*)d_in[7];
    const float* v_w2 = (const float*)d_in[8];  const float* v_b2 = (const float*)d_in[9];
    const float* t_w1 = (const float*)d_in[10]; const float* t_b1 = (const float*)d_in[11];
    const float* t_w2 = (const float*)d_in[12]; const float* t_b2 = (const float*)d_in[13];
    const float* pre_w1 = (const float*)d_in[14]; const float* pre_b1 = (const float*)d_in[15];
    const float* pre_w2 = (const float*)d_in[16]; const float* pre_b2 = (const float*)d_in[17];
    const float* s2v_in_w  = (const float*)d_in[18]; const float* s2v_in_b  = (const float*)d_in[19];
    const float* s2v_out_w = (const float*)d_in[20]; const float* s2v_out_b = (const float*)d_in[21];
    const float* qa2s_in_w  = (const float*)d_in[22]; const float* qa2s_in_b  = (const float*)d_in[23];
    const float* qa2s_out_w = (const float*)d_in[24]; const float* qa2s_out_b = (const float*)d_in[25];
    const float* gru_w_ih = (const float*)d_in[26]; const float* gru_w_hh = (const float*)d_in[27];
    const float* gru_b_ih = (const float*)d_in[28]; const float* gru_b_hh = (const float*)d_in[29];
    const float* proj_w1 = (const float*)d_in[30]; const float* proj_b1 = (const float*)d_in[31];
    const float* proj_w2 = (const float*)d_in[32]; const float* proj_b2 = (const float*)d_in[33];
    float* out = (float*)d_out;

    // ---- workspace bump allocator ----
    char* base = (char*)d_ws; size_t off = 0;
    auto alloc = [&](size_t bytes) -> void* {
        void* p = base + off; off = (off + bytes + 255) & ~(size_t)255; return p;
    };
    auto falloc = [&](long n) -> float* { return (float*)alloc((size_t)n * 4); };

    unsigned short* w1bf = (unsigned short*)alloc((size_t)cDIN * cDIN * 2);
    unsigned short* w2bf = (unsigned short*)alloc((size_t)cDS * cDIN * 2);
    float* vid_mid = falloc((long)cB * cV * cD);
    float* vidM    = falloc((long)cB * cV * cD);
    float* qp      = falloc((long)cB * cS * cD);
    float* kp      = falloc((long)cB * cV * cD);
    float* vp      = falloc((long)cB * cV * cD);
    float* scores  = falloc((long)cB * cNH * cS * cV);
    float* ctx     = falloc((long)cB * cS * cD);
    float* vid_att = falloc((long)cB * cS * cD);
    float* scr_mid = falloc((long)cB * cS * cD);
    float* scr_t   = falloc((long)cB * cS * cD);
    float* q_mid   = falloc((long)cB * cD);
    float* q_t     = falloc((long)cB * cD);
    float* kp2     = falloc((long)cB * cS * cD);
    float* vp2     = falloc((long)cB * cS * cD);
    float* at_mid  = falloc((long)cB * cA * cD);
    float* a_t     = falloc((long)cB * cA * cD);
    float* ab_mid  = falloc((long)cB * cA * cM * cD);
    float* ab_buf  = falloc((long)cB * cA * cM * cD);
    float* inputs  = falloc((long)cB * cA * cDIN);
    float* qp2     = falloc((long)cB * cA * cD);
    float* scores2 = falloc((long)cB * cNH * cA * cS);
    float* wmask   = falloc((long)cB * cA * cS);
    float* ctx2    = falloc((long)cB * cA * cD);
    float* pre_mid = falloc((long)cB * cA * cDIN);
    float* pre     = falloc((long)cB * cA * cDS);
    float* gi      = falloc((long)cB * cA * 3 * cDS);
    float* gh      = falloc((long)cB * 3 * cDS);
    float* states  = falloc((long)cB * cA * cDS);
    float* proj_mid= falloc((long)cB * cA * cDS);
    float* state   = falloc((long)cB * cDS);

    // ---- one-time-per-call: big weights to bf16 (L2-resident for the scan) ----
    k_f32_to_bf16<<<gcover((long)cDIN * cDIN), 256, 0, stream>>>(pre_w1, w1bf, (long)cDIN * cDIN);
    k_f32_to_bf16<<<gcover((long)cDS * cDIN), 256, 0, stream>>>(pre_w2, w2bf, (long)cDS * cDIN);

    // ---- precompute (vmapped over B via flattened rows) ----
    gemm(stream, video, cD, 0, 0, v_w1, false, cD, 0, 0, 0,
         vid_mid, cD, 0, 0, v_b1, cB * cV, cD, cD, 1, 1, 1.0f, 1);
    gemm(stream, vid_mid, cD, 0, 0, v_w2, false, cD, 0, 0, 0,
         vidM, cD, 0, 0, v_b2, cB * cV, cD, cD, 1, 1, 1.0f, 0);
    gemm(stream, script, cD, 0, 0, s2v_in_w, false, cD, 0, 0, 0,
         qp, cD, 0, 0, s2v_in_b, cB * cS, cD, cD, 1, 1, 1.0f, 0);
    gemm(stream, vidM, cD, 0, 0, s2v_in_w + (long)cD * cD, false, cD, 0, 0, 0,
         kp, cD, 0, 0, s2v_in_b + cD, cB * cV, cD, cD, 1, 1, 1.0f, 0);
    gemm(stream, vidM, cD, 0, 0, s2v_in_w + 2L * cD * cD, false, cD, 0, 0, 0,
         vp, cD, 0, 0, s2v_in_b + 2 * cD, cB * cV, cD, cD, 1, 1, 1.0f, 0);
    gemm(stream, qp, cD, (long)cS * cD, 64,
         kp, false, cD, (long)cV * cD, 64, 0,
         scores, cV, (long)cNH * cS * cV, (long)cS * cV,
         nullptr, cS, cV, 64, cB * cNH, cNH, 0.125f, 0);
    k_softmax_rows<<<cB * cNH * cS, 128, 0, stream>>>(scores, cV);
    gemm(stream, scores, cV, (long)cNH * cS * cV, (long)cS * cV,
         vp, false, cD, (long)cV * cD, 64, 1,
         ctx, cD, (long)cS * cD, 64,
         nullptr, cS, 64, cV, cB * cNH, cNH, 1.0f, 0);
    gemm(stream, ctx, cD, 0, 0, s2v_out_w, false, cD, 0, 0, 0,
         vid_att, cD, 0, 0, s2v_out_b, cB * cS, cD, cD, 1, 1, 1.0f, 0);
    gemm(stream, script, cD, 0, 0, t_w1, false, cD, 0, 0, 0,
         scr_mid, cD, 0, 0, t_b1, cB * cS, cD, cD, 1, 1, 1.0f, 1);
    gemm(stream, scr_mid, cD, 0, 0, t_w2, false, cD, 0, 0, 0,
         scr_t, cD, 0, 0, t_b2, cB * cS, cD, cD, 1, 1, 1.0f, 0);
    gemm(stream, question, cD, 0, 0, t_w1, false, cD, 0, 0, 0,
         q_mid, cD, 0, 0, t_b1, cB, cD, cD, 1, 1, 1.0f, 1);
    gemm(stream, q_mid, cD, 0, 0, t_w2, false, cD, 0, 0, 0,
         q_t, cD, 0, 0, t_b2, cB, cD, cD, 1, 1, 1.0f, 0);
    gemm(stream, scr_t, cD, 0, 0, qa2s_in_w + (long)cD * cD, false, cD, 0, 0, 0,
         kp2, cD, 0, 0, qa2s_in_b + cD, cB * cS, cD, cD, 1, 1, 1.0f, 0);
    gemm(stream, scr_t, cD, 0, 0, qa2s_in_w + 2L * cD * cD, false, cD, 0, 0, 0,
         vp2, cD, 0, 0, qa2s_in_b + 2 * cD, cB * cS, cD, cD, 1, 1, 1.0f, 0);

    k_bcast_state<<<gcover((long)cB * cDS), 256, 0, stream>>>(state0, state);

    // ---- sequential scan over T ----
    for (int t = 0; t < cT; ++t) {
        gemm(stream, a_texts + (long)t * cA * cD, cD, (long)cT * cA * cD, 0,
             t_w1, false, cD, 0, 0, 0,
             at_mid, cD, (long)cA * cD, 0, t_b1, cA, cD, cD, cB, 1, 1.0f, 1);
        gemm(stream, at_mid, cD, 0, 0, t_w2, false, cD, 0, 0, 0,
             a_t, cD, 0, 0, t_b2, cB * cA, cD, cD, 1, 1, 1.0f, 0);
        gemm(stream, a_buttons + (long)t * cA * cM * cD, cD, (long)cT * cA * cM * cD, 0,
             v_w1, false, cD, 0, 0, 0,
             ab_mid, cD, (long)cA * cM * cD, 0, v_b1, cA * cM, cD, cD, cB, 1, 1.0f, 1);
        gemm(stream, ab_mid, cD, 0, 0, v_w2, false, cD, 0, 0, 0,
             ab_buf, cD, 0, 0, v_b2, cB * cA * cM, cD, cD, 1, 1, 1.0f, 0);
        k_scatter_ab<<<gcover((long)cB * cA * cM * cD), 256, 0, stream>>>(ab_buf, inputs);
        k_qa_add<<<gcover((long)cB * cA * cD), 256, 0, stream>>>(q_t, a_t, inputs);
        gemm(stream, inputs + 2 * cD, cDIN, 0, 0, qa2s_in_w, false, cD, 0, 0, 0,
             qp2, cD, 0, 0, qa2s_in_b, cB * cA, cD, cD, 1, 1, 1.0f, 0);
        gemm(stream, qp2, cD, (long)cA * cD, 64,
             kp2, false, cD, (long)cS * cD, 64, 0,
             scores2, cS, (long)cNH * cA * cS, (long)cA * cS,
             nullptr, cA, cS, 64, cB * cNH, cNH, 0.125f, 0);
        k_softmax_rows<<<cB * cNH * cA, 128, 0, stream>>>(scores2, cS);
        k_wmask_mean<<<gcover((long)cB * cA * cS), 256, 0, stream>>>(scores2, wmask);
        gemm(stream, scores2, cS, (long)cNH * cA * cS, (long)cA * cS,
             vp2, false, cD, (long)cS * cD, 64, 1,
             ctx2, cD, (long)cA * cD, 64,
             nullptr, cA, 64, cS, cB * cNH, cNH, 1.0f, 0);
        gemm(stream, ctx2, cD, 0, 0, qa2s_out_w, false, cD, 0, 0, 0,
             inputs + cD, cDIN, 0, 0, qa2s_out_b, cB * cA, cD, cD, 1, 1, 1.0f, 0);
        gemm(stream, wmask, cS, (long)cA * cS, 0,
             vid_att, false, cD, (long)cS * cD, 0, 1,
             inputs, cDIN, (long)cA * cDIN, 0,
             nullptr, cA, cD, cS, cB, 1, 1.0f, 0);
        gemm(stream, inputs, cDIN, 0, 0, w1bf, true, cDIN, 0, 0, 0,
             pre_mid, cDIN, 0, 0, pre_b1, cB * cA, cDIN, cDIN, 1, 1, 1.0f, 1);
        gemm(stream, pre_mid, cDIN, 0, 0, w2bf, true, cDIN, 0, 0, 0,
             pre, cDS, 0, 0, pre_b2, cB * cA, cDS, cDIN, 1, 1, 1.0f, 0);
        gemm(stream, pre, cDS, 0, 0, gru_w_ih, false, cDS, 0, 0, 0,
             gi, 3 * cDS, 0, 0, gru_b_ih, cB * cA, 3 * cDS, cDS, 1, 1, 1.0f, 0);
        gemm(stream, state, cDS, 0, 0, gru_w_hh, false, cDS, 0, 0, 0,
             gh, 3 * cDS, 0, 0, gru_b_hh, cB, 3 * cDS, cDS, 1, 1, 1.0f, 0);
        k_gru<<<gcover((long)cB * cA * cDS), 256, 0, stream>>>(gi, gh, state, states);
        gemm(stream, states, cDS, 0, 0, proj_w1, false, cDS, 0, 0, 0,
             proj_mid, cDS, 0, 0, proj_b1, cB * cA, cDS, cDS, 1, 1, 1.0f, 1);
        k_logits_argmax<<<cB, 256, 0, stream>>>(proj_mid, proj_w2, proj_b2,
                                                pre, out, state, t);
    }
}
